// AAST_block_9302899163169
// MI455X (gfx1250) — matrix operand order repaired and checked by reading, not compiled, not run on hardware
//
#include <hip/hip_runtime.h>

typedef __bf16 bf16_t;
typedef __attribute__((ext_vector_type(16))) __bf16 v16bf;
typedef __attribute__((ext_vector_type(8)))  float  v8f;

#define B_   16
#define N_   1024
#define F_   32
#define T_   12
#define FC_  64
#define FT_  64
#define NN_  (N_*N_)
#define CDIM (F_*T_)   // 384

#define TM  128
#define TN  64
#define TK  64
#define SA  72         // LDS row stride for A tile (bf16 elems; 144B rows, 16B aligned)
#define SB  66         // LDS row stride for transposed B tile (bf16 elems)

#define WMMA_BF16(a, b, c) \
  __builtin_amdgcn_wmma_f32_16x16x32_bf16(false, (a), false, (b), (short)0, (c), false, false)

// ---------------- WMMA fragment helpers (16x16x32 bf16) ----------------
// A fragment: lane (half = l>>4, m = l&15) holds row m; K groups:
//   elems 0..7  -> K = half*8 + e ;  elems 8..15 -> K = 16 + half*8 + (e-8)
__device__ inline v16bf frag_a_row(const bf16_t* Ar, int half) {
  v16bf a;
#pragma unroll
  for (int i = 0; i < 4; ++i) {
    int k = half * 8 + 2 * i;
    a[2*i]     = Ar[k];
    a[2*i+1]   = Ar[k+1];
    a[8+2*i]   = Ar[16+k];
    a[8+2*i+1] = Ar[16+k+1];
  }
  return a;
}
// B fragment from transposed LDS tile Bt[n][k]: lane col n, K = half*8 + e (e<8), 16 + half*8 + (e-8) (e>=8)
__device__ inline v16bf frag_b_row(const bf16_t* Br, int half) {
  v16bf b;
#pragma unroll
  for (int e = 0; e < 16; ++e) b[e] = Br[(e < 8) ? (half*8 + e) : (16 + half*8 + (e - 8))];
  return b;
}

// ---------------- generic NN bf16 GEMM: C[z] = alpha*(A @ B[z]) + diag*I ---
// A tile staged via CDNA5 async global->LDS DMA (ASYNCcnt), B tile transposed
// through VGPRs. 8 waves x (2x2) 16x16 tiles, TK=64 -> 8 WMMA per wave/iter.
__global__ void __launch_bounds__(256)
k_gemm_nn(const bf16_t* __restrict__ A, const bf16_t* __restrict__ Bm,
          float* __restrict__ C, long strideB, long strideC,
          float alpha, float diag) {
  __shared__ bf16_t At[TM * SA];
  __shared__ bf16_t Bt[TN * SB];
  int z  = blockIdx.z;
  int m0 = blockIdx.x * TM, n0 = blockIdx.y * TN;
  int tid = threadIdx.x;
  int lane = tid & 31, wave = tid >> 5;
  int wm = wave & 3, wn = wave >> 2;
  int half = lane >> 4, mr = lane & 15;
  const bf16_t* Bz = Bm + (long)z * strideB;
  v8f acc[2][2] = {};
  for (int k0 = 0; k0 < N_; k0 += TK) {
    // A tile: 128 rows x 64 bf16 = 1024 x 16B async lane-copies
#pragma unroll
    for (int i = 0; i < 4; ++i) {
      int c = tid + i * 256;
      int row = c >> 3, seg = c & 7;
      const bf16_t* g = A + (long)(m0 + row) * N_ + k0 + seg * 8;
      unsigned l = (unsigned)(size_t)&At[row * SA + seg * 8];
      asm volatile("global_load_async_to_lds_b128 %0, %1, off"
                   :: "v"(l), "v"(g) : "memory");
    }
    // B tile transposed: Bt[nn][kk] = B[k0+kk][n0+nn]
#pragma unroll
    for (int i = 0; i < 16; ++i) {
      int c = tid + i * 256;
      int kk = c >> 6, nn = c & 63;
      Bt[nn * SB + kk] = Bz[(long)(k0 + kk) * N_ + n0 + nn];
    }
    asm volatile("s_wait_asynccnt 0" ::: "memory");
    __syncthreads();
#pragma unroll
    for (int ks = 0; ks < 2; ++ks) {
      const bf16_t* Ab = &At[(wm * 32 + mr) * SA + ks * 32];
      const bf16_t* Bb = &Bt[(wn * 32 + mr) * SB + ks * 32];
      v16bf a0 = frag_a_row(Ab, half);
      v16bf a1 = frag_a_row(Ab + 16 * SA, half);
      v16bf b0 = frag_b_row(Bb, half);
      v16bf b1 = frag_b_row(Bb + 16 * SB, half);
      acc[0][0] = WMMA_BF16(a0, b0, acc[0][0]);
      acc[0][1] = WMMA_BF16(a0, b1, acc[0][1]);
      acc[1][0] = WMMA_BF16(a1, b0, acc[1][0]);
      acc[1][1] = WMMA_BF16(a1, b1, acc[1][1]);
    }
    __syncthreads();
  }
  float* Cz = C + (long)z * strideC;
#pragma unroll
  for (int cm = 0; cm < 2; ++cm) {
#pragma unroll
    for (int cn = 0; cn < 2; ++cn) {
      int col = n0 + wn * 32 + cn * 16 + mr;
#pragma unroll
      for (int r = 0; r < 8; ++r) {
        int row = m0 + wm * 32 + cm * 16 + half * 8 + r;
        Cz[(long)row * N_ + col] =
            alpha * acc[cm][cn][r] + ((row == col) ? diag : 0.f);
      }
    }
  }
}

// -------- masked graph-conv GEMM: H[k,b] = (poly_k ⊙ spat_b)^T @ X_b ------
// A tile is a fused transform (f32 poly*spat -> bf16, transposed), so it is
// built through VGPRs; B tile converts x rows f32->bf16 transposed.
__global__ void __launch_bounds__(256)
k_gemm_H(const float* __restrict__ poly1, const float* __restrict__ poly2,
         const float* __restrict__ spat, const float* __restrict__ x,
         float* __restrict__ H) {
  __shared__ bf16_t At[TM * SA];
  __shared__ bf16_t Bt[TN * SB];
  int bz = blockIdx.z;
  int b = bz >> 1, k2 = bz & 1;
  const float* poly = k2 ? poly2 : poly1;
  int m0 = blockIdx.x * TM, c0 = blockIdx.y * TN;
  int tid = threadIdx.x;
  int lane = tid & 31, wave = tid >> 5;
  int wm = wave & 3, wn = wave >> 2;
  int half = lane >> 4, mr = lane & 15;
  const float* sb = spat + (long)b * NN_;
  const float* xb = x + (long)b * N_ * CDIM;
  v8f acc[2][2] = {};
  for (int n0 = 0; n0 < N_; n0 += TK) {
    // At[mi][kk] = poly[(n0+kk)*N + m0+mi] * spat[b][(n0+kk)*N + m0+mi]
#pragma unroll
    for (int i = 0; i < 4; ++i) {
      int c = tid + i * 256;
      int kk = c >> 4, mi0 = (c & 15) * 8;
      long off = (long)(n0 + kk) * N_ + m0 + mi0;
      const float* p = poly + off;
      const float* s = sb + off;
      if (n0 + TK < N_) __builtin_prefetch(sb + off + (long)TK * N_, 0, 1);
#pragma unroll
      for (int j = 0; j < 8; ++j)
        At[(mi0 + j) * SA + kk] = (bf16_t)(p[j] * s[j]);
    }
    // Bt[cc][kk] = x[b][n0+kk][c0+cc]   (c = f*T + t, contiguous per node)
#pragma unroll
    for (int i = 0; i < 2; ++i) {
      int c = tid + i * 256;
      int kk = c >> 3, cc0 = (c & 7) * 8;
      const float* src = xb + (long)(n0 + kk) * CDIM + c0 + cc0;
#pragma unroll
      for (int j = 0; j < 8; ++j) Bt[(cc0 + j) * SB + kk] = (bf16_t)src[j];
    }
    __syncthreads();
#pragma unroll
    for (int ks = 0; ks < 2; ++ks) {
      const bf16_t* Ab = &At[(wm * 32 + mr) * SA + ks * 32];
      const bf16_t* Bb = &Bt[(wn * 32 + mr) * SB + ks * 32];
      v16bf a0 = frag_a_row(Ab, half);
      v16bf a1 = frag_a_row(Ab + 16 * SA, half);
      v16bf b0 = frag_b_row(Bb, half);
      v16bf b1 = frag_b_row(Bb + 16 * SB, half);
      acc[0][0] = WMMA_BF16(a0, b0, acc[0][0]);
      acc[0][1] = WMMA_BF16(a0, b1, acc[0][1]);
      acc[1][0] = WMMA_BF16(a1, b0, acc[1][0]);
      acc[1][1] = WMMA_BF16(a1, b1, acc[1][1]);
    }
    __syncthreads();
  }
  float* Hb = H + ((long)k2 * B_ + b) * N_ * CDIM;
#pragma unroll
  for (int cm = 0; cm < 2; ++cm) {
#pragma unroll
    for (int cn = 0; cn < 2; ++cn) {
      int col = c0 + wn * 32 + cn * 16 + mr;
#pragma unroll
      for (int r = 0; r < 8; ++r) {
        int row = m0 + wm * 32 + cm * 16 + half * 8 + r;
        Hb[(long)row * CDIM + col] = acc[cm][cn][r];
      }
    }
  }
}

// ---------------- small VALU kernels ----------------
__global__ void k_btf(const float* x, const float* U1, float* btf) {
  int idx = blockIdx.x * blockDim.x + threadIdx.x;
  if (idx >= B_ * T_ * F_) return;
  int f = idx % F_, t = (idx / F_) % T_, b = idx / (F_ * T_);
  float s = 0.f;
  for (int n = 0; n < N_; ++n) s += x[((long)(b * N_ + n) * F_ + f) * T_ + t] * U1[n];
  btf[idx] = s;
}
__global__ void k_lhs(const float* btf, const float* U2, float* lhs) {
  int idx = blockIdx.x * blockDim.x + threadIdx.x;
  if (idx >= B_ * T_ * N_) return;
  int n = idx % N_, t = (idx / N_) % T_, b = idx / (N_ * T_);
  float s = 0.f;
  for (int f = 0; f < F_; ++f) s += btf[(b * T_ + t) * F_ + f] * U2[f * N_ + n];
  lhs[idx] = s;
}
__global__ void k_rhs(const float* x, const float* U3, float* rhs) {
  int idx = blockIdx.x * blockDim.x + threadIdx.x;
  if (idx >= B_ * N_ * T_) return;
  int t = idx % T_, n = (idx / T_) % N_, b = idx / (T_ * N_);
  float s = 0.f;
  for (int f = 0; f < F_; ++f) s += U3[f] * x[((long)(b * N_ + n) * F_ + f) * T_ + t];
  rhs[idx] = s;
}
__global__ void k_E(const float* lhs, const float* rhs, const float* be,
                    const float* Ve, float* E) {
  int b = blockIdx.x, tid = threadIdx.x;
  __shared__ float Ps[T_ * T_];
  __shared__ float E1[T_ * T_];
  if (tid < T_ * T_) {
    int u = tid % T_, t = tid / T_;
    float s = 0.f;
    for (int n = 0; n < N_; ++n)
      s += lhs[(b * T_ + t) * N_ + n] * rhs[(b * N_ + n) * T_ + u];
    s += be[t * T_ + u];
    Ps[tid] = 1.f / (1.f + __expf(-s));
  }
  __syncthreads();
  if (tid < T_ * T_) {
    int u = tid % T_, t = tid / T_;
    float s = 0.f;
    for (int k = 0; k < T_; ++k) s += Ve[t * T_ + k] * Ps[k * T_ + u];
    E1[tid] = s;
  }
  __syncthreads();
  if (tid < T_) { // softmax over axis t for column u = tid
    int u = tid;
    float m = -1e30f;
    for (int t = 0; t < T_; ++t) m = fmaxf(m, E1[t * T_ + u]);
    float den = 0.f;
    for (int t = 0; t < T_; ++t) den += __expf(E1[t * T_ + u] - m);
    for (int t = 0; t < T_; ++t)
      E[(b * T_ + t) * T_ + u] = __expf(E1[t * T_ + u] - m) / den;
  }
}
__global__ void k_xTAt(const float* x, const float* E, float* xT) {
  long idx = (long)blockIdx.x * blockDim.x + threadIdx.x;
  if (idx >= (long)B_ * N_ * F_ * T_) return;
  int u = idx % T_; long r = idx / T_;
  int b = (int)(r / (N_ * F_));
  const float* xr = &x[r * T_];
  float s = 0.f;
  for (int t = 0; t < T_; ++t) s += xr[t] * E[(b * T_ + t) * T_ + u];
  xT[idx] = s;
}
__global__ void k_l2a(const float* xT, const float* W1, float* l2a) {
  long idx = (long)blockIdx.x * blockDim.x + threadIdx.x;
  if (idx >= (long)B_ * N_ * F_) return;
  const float* r = &xT[idx * T_];
  float s = 0.f;
  for (int t = 0; t < T_; ++t) s += r[t] * W1[t];
  l2a[idx] = s;
}
__global__ void k_l2(const float* l2a, const float* W2, float* l2) {
  int idx = blockIdx.x * blockDim.x + threadIdx.x;
  if (idx >= B_ * N_ * T_) return;
  int t = idx % T_; long bn = idx / T_;
  float s = 0.f;
  for (int f = 0; f < F_; ++f) s += l2a[bn * F_ + f] * W2[f * T_ + t];
  l2[idx] = s;
}
__global__ void k_r2(const float* xT, const float* W3, float* r2) {
  int idx = blockIdx.x * blockDim.x + threadIdx.x;
  if (idx >= B_ * T_ * N_) return;
  int n = idx % N_, t = (idx / N_) % T_, b = idx / (N_ * T_);
  float s = 0.f;
  for (int f = 0; f < F_; ++f) s += W3[f] * xT[((long)(b * N_ + n) * F_ + f) * T_ + t];
  r2[idx] = s;
}
__global__ void k_sig(const float* l2, const float* r2, const float* bs, bf16_t* sig) {
  long idx = (long)blockIdx.x * blockDim.x + threadIdx.x;
  if (idx >= (long)B_ * NN_) return;
  int j = idx % N_; int i = (int)((idx / N_) % N_); int b = (int)(idx / NN_);
  float s = 0.f;
  for (int t = 0; t < T_; ++t) s += l2[(b * N_ + i) * T_ + t] * r2[(b * T_ + t) * N_ + j];
  s += bs[i * N_ + j];
  sig[idx] = (bf16_t)(1.f / (1.f + __expf(-s)));
}
__global__ void k_cvt(const float* src, bf16_t* dst, long n) {
  long i = (long)blockIdx.x * blockDim.x + threadIdx.x;
  if (i < n) dst[i] = (bf16_t)src[i];
}
// softmax over axis=1 (stride N_) for fixed (b,k); apply sparsity threshold.
// mean(Sn) == 1/N analytically (each softmax column sums to 1).
__global__ void __launch_bounds__(256) k_softmax_col_thr(float* S) {
  int b = blockIdx.x / N_, k = blockIdx.x % N_;
  float* col = S + (long)b * NN_ + k;
  __shared__ float red[256];
  int tid = threadIdx.x;
  float v[4]; float m = -1e30f;
  for (int i = 0; i < 4; ++i) { v[i] = col[(long)(tid + i * 256) * N_]; m = fmaxf(m, v[i]); }
  red[tid] = m; __syncthreads();
  for (int s = 128; s > 0; s >>= 1) { if (tid < s) red[tid] = fmaxf(red[tid], red[tid + s]); __syncthreads(); }
  m = red[0]; __syncthreads();
  float sum = 0.f;
  for (int i = 0; i < 4; ++i) { v[i] = __expf(v[i] - m); sum += v[i]; }
  red[tid] = sum; __syncthreads();
  for (int s = 128; s > 0; s >>= 1) { if (tid < s) red[tid] += red[tid + s]; __syncthreads(); }
  sum = red[0];
  const float thr = 1.0f / (0.6f * (float)N_);
  for (int i = 0; i < 4; ++i) {
    float val = v[i] / sum;
    col[(long)(tid + i * 256) * N_] = (val < thr) ? 0.f : val;
  }
}
__global__ void k_G(const float* NE, float* G) {
  int idx = blockIdx.x * blockDim.x + threadIdx.x;
  if (idx >= NN_) return;
  int m = idx % N_, n = idx / N_;
  float s = 0.f;
  for (int d = 0; d < 16; ++d) s += NE[n * 16 + d] * NE[m * 16 + d];
  G[idx] = fmaxf(s, 0.f);
}
__global__ void __launch_bounds__(256) k_row_softmax(float* M) {
  float* row = M + (long)blockIdx.x * N_;
  __shared__ float red[256];
  int tid = threadIdx.x;
  float v[4]; float m = -1e30f;
  for (int i = 0; i < 4; ++i) { v[i] = row[tid + i * 256]; m = fmaxf(m, v[i]); }
  red[tid] = m; __syncthreads();
  for (int s = 128; s > 0; s >>= 1) { if (tid < s) red[tid] = fmaxf(red[tid], red[tid + s]); __syncthreads(); }
  m = red[0]; __syncthreads();
  float sum = 0.f;
  for (int i = 0; i < 4; ++i) { v[i] = __expf(v[i] - m); sum += v[i]; }
  red[tid] = sum; __syncthreads();
  for (int s = 128; s > 0; s >>= 1) { if (tid < s) red[tid] += red[tid + s]; __syncthreads(); }
  sum = red[0];
  for (int i = 0; i < 4; ++i) row[tid + i * 256] = v[i] / sum;
}
__global__ void k_gcn(const float* x, const float* spat, const float* H,
                      const float* Theta, float* gcn) {
  long idx = (long)blockIdx.x * blockDim.x + threadIdx.x;
  if (idx >= (long)B_ * T_ * N_ * FC_) return;
  int o = idx % FC_; int m = (int)((idx / FC_) % N_);
  int t = (int)((idx / ((long)FC_ * N_)) % T_); int b = (int)(idx / ((long)FC_ * N_ * T_));
  const float* xm = &x[((long)(b * N_ + m) * F_) * T_ + t];
  float diag = spat[(long)b * NN_ + (long)m * N_ + m];
  const float* H1 = &H[((long)b * N_ + m) * CDIM + t];
  const float* H2 = &H[(((long)B_ + b) * N_ + m) * CDIM + t];
  float s = 0.f;
  for (int f = 0; f < F_; ++f) {
    s += Theta[(0 * F_ + f) * FC_ + o] * diag * xm[f * T_];
    s += Theta[(1 * F_ + f) * FC_ + o] * H1[f * T_];
    s += Theta[(2 * F_ + f) * FC_ + o] * H2[f * T_];
  }
  gcn[idx] = fmaxf(s, 0.f);
}
__global__ void k_wnorm(const float* tc_v, const float* tc_g, float* w) {
  int co = blockIdx.x * blockDim.x + threadIdx.x;
  if (co >= FT_) return;
  float ss = 0.f;
  for (int i = 0; i < FC_ * 3; ++i) { float v = tc_v[co * FC_ * 3 + i]; ss += v * v; }
  float sc = tc_g[co] * rsqrtf(ss);
  for (int i = 0; i < FC_ * 3; ++i) w[co * FC_ * 3 + i] = tc_v[co * FC_ * 3 + i] * sc;
}
__global__ void __launch_bounds__(256)
k_final(const float* x, const float* gcn, const float* w, const float* tc_b,
        const float* rc_w, const float* rc_b, const float* ln_w,
        const float* ln_b, float* out) {
  int b = blockIdx.x / N_, n = blockIdx.x % N_;
  __shared__ float g[T_ * FC_];
  __shared__ float xr[F_ * T_];
  __shared__ float y[T_ * FT_];
  int tid = threadIdx.x;
  for (int i = tid; i < T_ * FC_; i += 256) {
    int t = i / FC_, ci = i % FC_;
    g[i] = gcn[(((long)b * T_ + t) * N_ + n) * FC_ + ci];
  }
  for (int i = tid; i < F_ * T_; i += 256)
    xr[i] = x[(long)(b * N_ + n) * F_ * T_ + i];
  __syncthreads();
  for (int i = tid; i < T_ * FT_; i += 256) {
    int t = i / FT_, c = i % FT_;
    float s = tc_b[c] + rc_b[c];
    for (int kt = 0; kt < 3; ++kt) {
      int tt = t + kt - 1;
      if (tt < 0 || tt >= T_) continue;
      const float* wc = &w[c * FC_ * 3 + kt];
      for (int ci = 0; ci < FC_; ++ci) s += wc[ci * 3] * g[tt * FC_ + ci];
    }
    for (int f = 0; f < F_; ++f) s += rc_w[c * F_ + f] * xr[f * T_ + t];
    y[t * FT_ + c] = s;
  }
  __syncthreads();
  if (tid < T_) {
    int t = tid;
    float mu = 0.f;
    for (int c = 0; c < FT_; ++c) mu += y[t * FT_ + c];
    mu /= FT_;
    float var = 0.f;
    for (int c = 0; c < FT_; ++c) { float d = y[t * FT_ + c] - mu; var += d * d; }
    var /= FT_;
    float inv = rsqrtf(var + 1e-5f);
    for (int c = 0; c < FT_; ++c) {
      float val = (y[t * FT_ + c] - mu) * inv * ln_w[c] + ln_b[c];
      out[(((long)b * N_ + n) * FT_ + c) * T_ + t] = val;
    }
  }
}

// ---------------- host launcher ----------------
static inline int blks(long n) { return (int)((n + 255) / 256); }

extern "C" void kernel_launch(void* const* d_in, const int* in_sizes, int n_in,
                              void* d_out, int out_size, void* d_ws, size_t ws_size,
                              hipStream_t stream) {
  const float* x   = (const float*)d_in[0];
  const float* NE  = (const float*)d_in[1];
  const float* W1  = (const float*)d_in[2];
  const float* W2  = (const float*)d_in[3];
  const float* W3  = (const float*)d_in[4];
  const float* bs  = (const float*)d_in[5];
  const float* Vs  = (const float*)d_in[6];
  const float* U1  = (const float*)d_in[7];
  const float* U2  = (const float*)d_in[8];
  const float* U3  = (const float*)d_in[9];
  const float* be  = (const float*)d_in[10];
  const float* Ve  = (const float*)d_in[11];
  const float* Th  = (const float*)d_in[12];
  const float* tcv = (const float*)d_in[13];
  const float* tcg = (const float*)d_in[14];
  const float* tcb = (const float*)d_in[15];
  const float* rcw = (const float*)d_in[16];
  const float* rcb = (const float*)d_in[17];
  const float* lnw = (const float*)d_in[18];
  const float* lnb = (const float*)d_in[19];

  float* out_xres = (float*)d_out;                       // B*N*FT*T floats
  float* spat     = out_xres + (long)B_ * N_ * FT_ * T_; // B*N*N floats

  char* ws = (char*)d_ws;
  // region A: xTAt (25.1MB) + sig (33.5MB), later overlaid by gcn (50.3MB)
  float*  xT  = (float*)(ws + 0);
  bf16_t* sig = (bf16_t*)(ws + 25165824);
  float*  gcn = (float*)(ws + 0);
  size_t off = 58720256;
  auto alloc = [&](size_t bytes) { size_t r = off; off = (off + bytes + 255) & ~(size_t)255; return r; };
  float*  btf   = (float*)(ws + alloc(6144 * 4));
  float*  lhs   = (float*)(ws + alloc((size_t)B_ * T_ * N_ * 4));
  float*  rhs   = (float*)(ws + alloc((size_t)B_ * N_ * T_ * 4));
  float*  E     = (float*)(ws + alloc((size_t)B_ * T_ * T_ * 4));
  float*  l2a   = (float*)(ws + alloc((size_t)B_ * N_ * F_ * 4));
  float*  l2    = (float*)(ws + alloc((size_t)B_ * N_ * T_ * 4));
  float*  r2    = (float*)(ws + alloc((size_t)B_ * T_ * N_ * 4));
  bf16_t* Vsb   = (bf16_t*)(ws + alloc((size_t)NN_ * 2));
  float*  sup   = (float*)(ws + alloc((size_t)NN_ * 4));
  bf16_t* supb  = (bf16_t*)(ws + alloc((size_t)NN_ * 2));
  float*  poly2 = (float*)(ws + alloc((size_t)NN_ * 4));
  float*  H     = (float*)(ws + alloc((size_t)2 * B_ * N_ * CDIM * 4));
  float*  wtc   = (float*)(ws + alloc((size_t)FT_ * FC_ * 3 * 4));

  // --- temporal attention ---
  k_btf<<<blks(B_ * T_ * F_), 256, 0, stream>>>(x, U1, btf);
  k_lhs<<<blks(B_ * T_ * N_), 256, 0, stream>>>(btf, U2, lhs);
  k_rhs<<<blks(B_ * N_ * T_), 256, 0, stream>>>(x, U3, rhs);
  k_E<<<B_, 256, 0, stream>>>(lhs, rhs, be, Ve, E);
  k_xTAt<<<blks((long)B_ * N_ * F_ * T_), 256, 0, stream>>>(x, E, xT);

  // --- spatial attention ---
  k_l2a<<<blks((long)B_ * N_ * F_), 256, 0, stream>>>(xT, W1, l2a);
  k_l2<<<blks(B_ * N_ * T_), 256, 0, stream>>>(l2a, W2, l2);
  k_r2<<<blks(B_ * T_ * N_), 256, 0, stream>>>(xT, W3, r2);
  k_sig<<<blks((long)B_ * NN_), 256, 0, stream>>>(l2, r2, bs, sig);
  k_cvt<<<blks(NN_), 256, 0, stream>>>(Vs, Vsb, NN_);
  k_gemm_nn<<<dim3(N_ / TM, N_ / TN, B_), 256, 0, stream>>>(
      Vsb, sig, spat, (long)NN_, (long)NN_, 1.f, 0.f);
  k_softmax_col_thr<<<B_ * N_, 256, 0, stream>>>(spat);

  // --- Chebyshev supports ---
  k_G<<<blks(NN_), 256, 0, stream>>>(NE, sup);
  k_row_softmax<<<N_, 256, 0, stream>>>(sup);
  k_cvt<<<blks(NN_), 256, 0, stream>>>(sup, supb, NN_);
  k_gemm_nn<<<dim3(N_ / TM, N_ / TN, 1), 256, 0, stream>>>(
      supb, supb, poly2, 0L, 0L, 2.f, -1.f);

  // --- masked graph conv (dominant GEMM) ---
  k_gemm_H<<<dim3(N_ / TM, CDIM / TN, B_ * 2), 256, 0, stream>>>(
      sup, poly2, spat, x, H);
  k_gcn<<<blks((long)B_ * T_ * N_ * FC_), 256, 0, stream>>>(x, spat, H, Th, gcn);

  // --- time conv + residual + layernorm ---
  k_wnorm<<<1, 64, 0, stream>>>(tcv, tcg, wtc);
  k_final<<<B_ * N_, 256, 0, stream>>>(x, gcn, wtc, tcb, rcw, rcb, lnw, lnb, out_xres);
}
